// GroupedQueryAttention_10522669875724
// MI455X (gfx1250) — matrix operand order repaired and checked by reading, not compiled, not run on hardware
//
#include <hip/hip_runtime.h>

// ---------------------------------------------------------------------------
// Types for CDNA5 WMMA (wave32): v16bf = 16 x bf16 (8 VGPRs), v8f = f32 accum
// ---------------------------------------------------------------------------
typedef __attribute__((ext_vector_type(16))) __bf16 v16bf;
typedef __attribute__((ext_vector_type(8)))  float  v8f;

union V16 { v16bf v; uint4 q[2]; };

__device__ __forceinline__ unsigned short f2bf(float f) {
    unsigned int u = __builtin_bit_cast(unsigned int, f);
    u += 0x7FFFu + ((u >> 16) & 1u);   // round-to-nearest-even
    return (unsigned short)(u >> 16);
}

// CDNA5 async global->LDS copy (ASYNCcnt path). LDS operand = low 32 bits of
// the generic pointer to __shared__ (hardware adds wave LDS_BASE).
__device__ __forceinline__ void async_ld_b128(const unsigned short* g,
                                              unsigned short* l) {
    unsigned loff = (unsigned)(unsigned long long)l;
    asm volatile("global_load_async_to_lds_b128 %0, %1, off"
                 :: "v"(loff), "v"(g) : "memory");
}

// ---------------------------------------------------------------------------
// f32 -> bf16 elementwise
// ---------------------------------------------------------------------------
__global__ void k_f32_to_bf16(const float* __restrict__ src,
                              unsigned short* __restrict__ dst, int n) {
    int i = blockIdx.x * blockDim.x + threadIdx.x;
    if (i < n) dst[i] = f2bf(src[i]);
}

// ---------------------------------------------------------------------------
// Build transposed bf16 weights:
//   Wqkvt: (3072 x 2048), rows 0..2047 = Wq^T, 2048..2559 = Wk^T, rest Wv^T
//   Wot:   (2048 x 2048) = Wo^T
// ---------------------------------------------------------------------------
__global__ void k_pack_weights(const float* __restrict__ Wq, const float* __restrict__ Wk,
                               const float* __restrict__ Wv, const float* __restrict__ Wo,
                               unsigned short* __restrict__ Wqkvt,
                               unsigned short* __restrict__ Wot) {
    const int NQK = 3072 * 2048;
    int i = blockIdx.x * blockDim.x + threadIdx.x;
    if (i < NQK) {
        int n = i >> 11, k = i & 2047;
        float v;
        if (n < 2048)      v = Wq[k * 2048 + n];
        else if (n < 2560) v = Wk[k * 512 + (n - 2048)];
        else               v = Wv[k * 512 + (n - 2560)];
        Wqkvt[i] = f2bf(v);
    } else {
        int j = i - NQK;
        if (j < 2048 * 2048) {
            int n = j >> 11, k = j & 2047;
            Wot[j] = f2bf(Wo[k * 2048 + n]);
        }
    }
}

// ---------------------------------------------------------------------------
// bf16 GEMM: C(MxN,f32) = A(MxK, row-major bf16) * Bt(NxK, row-major bf16)^T
// Block tile 128x128, K-step 32, 8 waves; each wave 32x64 (2x4 wmma tiles).
// Double-buffered LDS filled by async global->LDS DMA (ASYNCcnt).
// ---------------------------------------------------------------------------
__global__ __launch_bounds__(256) void k_gemm_bf16(const unsigned short* __restrict__ A,
                                                   const unsigned short* __restrict__ Bt,
                                                   float* __restrict__ C,
                                                   int M, int N, int K) {
    __shared__ unsigned short As[2][128][40];   // +8 pad vs 64 LDS banks
    __shared__ unsigned short Bs[2][128][40];
    const int tid  = threadIdx.x;
    const int lane = tid & 31, wid = tid >> 5;
    const int m0 = blockIdx.y * 128, n0 = blockIdx.x * 128;
    const int wm = (wid & 3) * 32, wn = (wid >> 2) * 64;

    v8f zero = {0.f,0.f,0.f,0.f,0.f,0.f,0.f,0.f};
    v8f acc[2][4];
#pragma unroll
    for (int i = 0; i < 2; i++)
#pragma unroll
        for (int j = 0; j < 4; j++) acc[i][j] = zero;

    const int lrow = tid >> 1, lhalf = (tid & 1) * 16;
    const int ml = lane & 15, hs = lane >> 4;
    const unsigned short* arow = A  + (size_t)(m0 + lrow) * K + lhalf;
    const unsigned short* brow = Bt + (size_t)(n0 + lrow) * K + lhalf;

    // prologue: async-fill buffer 0 with the first K tile
    async_ld_b128(arow,     &As[0][lrow][lhalf]);
    async_ld_b128(arow + 8, &As[0][lrow][lhalf + 8]);
    async_ld_b128(brow,     &Bs[0][lrow][lhalf]);
    async_ld_b128(brow + 8, &Bs[0][lrow][lhalf + 8]);

    for (int k0 = 0; k0 < K; k0 += 32) {
        const int buf = (k0 >> 5) & 1;
        if (k0 + 32 < K) {                      // kick off next tile -> other buffer
            const unsigned short* ag = arow + k0 + 32;
            const unsigned short* bg = brow + k0 + 32;
            async_ld_b128(ag,     &As[buf ^ 1][lrow][lhalf]);
            async_ld_b128(ag + 8, &As[buf ^ 1][lrow][lhalf + 8]);
            async_ld_b128(bg,     &Bs[buf ^ 1][lrow][lhalf]);
            async_ld_b128(bg + 8, &Bs[buf ^ 1][lrow][lhalf + 8]);
            // async loads complete in order: <=4 outstanding => current buffer done
            asm volatile("s_wait_asynccnt 0x4" ::: "memory");
        } else {
            asm volatile("s_wait_asynccnt 0x0" ::: "memory");
        }
        __syncthreads();                        // all waves' DMA for buf landed

        V16 af[2], bf[4];
        const int c0 = hs * 8;                  // A chunk offsets: {c0, c0+16}
#pragma unroll
        for (int sm = 0; sm < 2; sm++) {
            const unsigned short* p = &As[buf][wm + sm * 16 + ml][c0];
            af[sm].q[0] = *(const uint4*)p;
            af[sm].q[1] = *(const uint4*)(p + 16);
        }
        const int cb = hs * 8;                  // B chunk offsets: {cb, cb+16}
#pragma unroll
        for (int sn = 0; sn < 4; sn++) {
            const unsigned short* p = &Bs[buf][wn + sn * 16 + ml][cb];
            bf[sn].q[0] = *(const uint4*)p;
            bf[sn].q[1] = *(const uint4*)(p + 16);
        }
#pragma unroll
        for (int sm = 0; sm < 2; sm++)
#pragma unroll
            for (int sn = 0; sn < 4; sn++)
                acc[sm][sn] = __builtin_amdgcn_wmma_f32_16x16x32_bf16(
                    false, af[sm].v, false, bf[sn].v, (short)0, acc[sm][sn],
                    false, false);
        __syncthreads();                        // buf free before it is re-filled
    }

    // C/D layout: VGPR r -> row r + (lane>=16)*8, col = lane&15
#pragma unroll
    for (int sm = 0; sm < 2; sm++)
#pragma unroll
        for (int sn = 0; sn < 4; sn++)
#pragma unroll
            for (int r = 0; r < 8; r++) {
                int row = m0 + wm + sm * 16 + hs * 8 + r;
                int col = n0 + wn + sn * 16 + ml;
                C[(size_t)row * N + col] = acc[sm][sn][r];
            }
}

// ---------------------------------------------------------------------------
// RMSNorm + RoPE on Q and K rows (one wave per (b,t,head) row; D=128 = 32x4)
// qkv layout: row b*T+t, cols [0,2048)=Q heads, [2048,2560)=K groups, rest V
// ---------------------------------------------------------------------------
__global__ __launch_bounds__(256) void k_normrope(const float* __restrict__ qkv,
                                                  const float* __restrict__ cosp,
                                                  const float* __restrict__ sinp,
                                                  const float* __restrict__ qscale,
                                                  const float* __restrict__ kscale,
                                                  unsigned short* __restrict__ Qr,
                                                  unsigned short* __restrict__ Kr) {
    const int lane = threadIdx.x & 31, wid = threadIdx.x >> 5;
    const int ridx = blockIdx.x * 8 + wid;                 // < B*T*(H+G)=81920
    const int b = ridx / (2048 * 20);
    const int rem = ridx % (2048 * 20);
    const int t = rem / 20, hh = rem % 20;

    const float* src = qkv + (size_t)(b * 2048 + t) * 3072 +
                       (hh < 16 ? hh * 128 : 2048 + (hh - 16) * 128);
    const int d0 = lane * 4;
    float4 xv = *(const float4*)(src + d0);
    float ss = xv.x*xv.x + xv.y*xv.y + xv.z*xv.z + xv.w*xv.w;
#pragma unroll
    for (int off = 16; off > 0; off >>= 1) ss += __shfl_xor(ss, off, 32);
    float inv = rsqrtf(ss * (1.0f / 128.0f) + 1e-6f);

    const float* sc = (hh < 16) ? qscale : kscale;
    float4 sv = *(const float4*)(sc + d0);
    float xn[4] = { xv.x*inv*sv.x, xv.y*inv*sv.y, xv.z*inv*sv.z, xv.w*inv*sv.w };

    float4 cv  = *(const float4*)(cosp + t * 128 + d0);
    float4 snv = *(const float4*)(sinp + t * 128 + d0);
    float rot[4];
#pragma unroll
    for (int j = 0; j < 4; j++) {                 // partner at d +/- 64
        float p = __shfl_xor(xn[j], 16, 32);
        rot[j] = (lane < 16) ? -p : p;
    }
    float o0 = xn[0]*cv.x + rot[0]*snv.x;
    float o1 = xn[1]*cv.y + rot[1]*snv.y;
    float o2 = xn[2]*cv.z + rot[2]*snv.z;
    float o3 = xn[3]*cv.w + rot[3]*snv.w;

    unsigned int lo = (unsigned)f2bf(o0) | ((unsigned)f2bf(o1) << 16);
    unsigned int hi = (unsigned)f2bf(o2) | ((unsigned)f2bf(o3) << 16);
    unsigned short* dst = (hh < 16)
        ? Qr + ((size_t)((b * 16 + hh) * 2048 + t)) * 128 + d0
        : Kr + ((size_t)((b * 4 + (hh - 16)) * 2048 + t)) * 128 + d0;
    *(uint2*)dst = make_uint2(lo, hi);
}

// ---------------------------------------------------------------------------
// V transpose: qkv f32 V section -> bf16 Vt[b][g][d][t]
// ---------------------------------------------------------------------------
__global__ void k_vtrans(const float* __restrict__ qkv,
                         unsigned short* __restrict__ Vt) {
    int i = blockIdx.x * 256 + threadIdx.x;        // B*G*D*T = 2097152
    int t = i & 2047, d = (i >> 11) & 127, g = (i >> 18) & 3, b = i >> 20;
    float v = qkv[(size_t)(b * 2048 + t) * 3072 + 2560 + g * 128 + d];
    Vt[i] = f2bf(v);
}

// ---------------------------------------------------------------------------
// Flash attention: 4 waves x 16 q-rows per block, 32-key steps, causal.
// Q[b,h,t,d] bf16 (A frags), K[b,g,t,d] bf16 (B frags for QK^T),
// Vt[b,g,d,t] bf16 (B frags for P.V), out attnb[b,t,h*128+d] bf16.
// ---------------------------------------------------------------------------
__global__ __launch_bounds__(128) void k_attention(const unsigned short* __restrict__ Qr,
                                                   const unsigned short* __restrict__ Kr,
                                                   const unsigned short* __restrict__ Vt,
                                                   unsigned short* __restrict__ attnb) {
    __shared__ unsigned short Pls[4][16][40];      // per-wave P staging (C->A transpose)
    const int lane = threadIdx.x & 31, wid = threadIdx.x >> 5;
    const int blk = blockIdx.x;
    const int qb = blk & 31, h = (blk >> 5) & 15, b = blk >> 9;
    const int g = h >> 2;
    const int qbase = qb * 64 + wid * 16;
    const int ml = lane & 15, hs = lane >> 4;

    const unsigned short* Qp = Qr + (size_t)((b * 16 + h) * 2048) * 128;
    const unsigned short* Kp = Kr + (size_t)((b * 4 + g) * 2048) * 128;
    const unsigned short* Vp = Vt + (size_t)((b * 4 + g) * 128) * 2048;

    V16 qa[4];
#pragma unroll
    for (int dc = 0; dc < 4; dc++) {
        const unsigned short* p = Qp + (size_t)(qbase + ml) * 128 + dc * 32 + hs * 8;
        qa[dc].q[0] = *(const uint4*)p;
        qa[dc].q[1] = *(const uint4*)(p + 16);
    }

    v8f zero = {0.f,0.f,0.f,0.f,0.f,0.f,0.f,0.f};
    v8f o[8]; float m_[8], l_[8];
#pragma unroll
    for (int i = 0; i < 8; i++) { o[i] = zero; m_[i] = -1e30f; l_[i] = 0.f; }

    const float scale = 0.088388347648318447f;     // 1/sqrt(128)
    const int kend = qbase + 16;

    for (int k0 = 0; k0 < kend; k0 += 32) {
        // ---- S = Q @ K^T (two 16x16 column tiles) ----
        v8f s[2];
#pragma unroll
        for (int j = 0; j < 2; j++) {
            v8f c = zero;
#pragma unroll
            for (int dc = 0; dc < 4; dc++) {
                V16 bk;
                const unsigned short* p =
                    Kp + (size_t)(k0 + j * 16 + ml) * 128 + dc * 32 + hs * 8;
                bk.q[0] = *(const uint4*)p;
                bk.q[1] = *(const uint4*)(p + 16);
                c = __builtin_amdgcn_wmma_f32_16x16x32_bf16(
                        false, qa[dc].v, false, bk.v, (short)0, c, false, false);
            }
            s[j] = c;
        }
        // ---- online softmax over 32 keys ----
        float corr[8];
#pragma unroll
        for (int r = 0; r < 8; r++) {
            const int qi = qbase + r + hs * 8;
            float s0 = (k0 + ml      <= qi) ? s[0][r] * scale : -1e30f;
            float s1 = (k0 + 16 + ml <= qi) ? s[1][r] * scale : -1e30f;
            float mx = fmaxf(s0, s1);
            mx = fmaxf(mx, __shfl_xor(mx, 1, 32));
            mx = fmaxf(mx, __shfl_xor(mx, 2, 32));
            mx = fmaxf(mx, __shfl_xor(mx, 4, 32));
            mx = fmaxf(mx, __shfl_xor(mx, 8, 32));
            float mnew = fmaxf(m_[r], mx);
            float cr = __expf(m_[r] - mnew);
            float p0 = __expf(s0 - mnew), p1 = __expf(s1 - mnew);
            float rs = p0 + p1;
            rs += __shfl_xor(rs, 1, 32);
            rs += __shfl_xor(rs, 2, 32);
            rs += __shfl_xor(rs, 4, 32);
            rs += __shfl_xor(rs, 8, 32);
            l_[r] = l_[r] * cr + rs;
            m_[r] = mnew; corr[r] = cr;
            s[0][r] = p0; s[1][r] = p1;
        }
#pragma unroll
        for (int dt = 0; dt < 8; dt++)
#pragma unroll
            for (int r = 0; r < 8; r++) o[dt][r] *= corr[r];

        // ---- P: C layout -> A layout through per-wave LDS tile ----
#pragma unroll
        for (int r = 0; r < 8; r++) {
            Pls[wid][r + hs * 8][ml]      = f2bf(s[0][r]);
            Pls[wid][r + hs * 8][16 + ml] = f2bf(s[1][r]);
        }
        V16 ap;
        {
            const unsigned short* p = &Pls[wid][ml][hs * 8];
            ap.q[0] = *(const uint4*)p;
            ap.q[1] = *(const uint4*)(p + 16);
        }
        // ---- O += P @ V (8 d-tiles of 16) ----
#pragma unroll
        for (int dt = 0; dt < 8; dt++) {
            V16 bv;
            const unsigned short* p =
                Vp + (size_t)(dt * 16 + ml) * 2048 + k0 + hs * 8;
            bv.q[0] = *(const uint4*)p;
            bv.q[1] = *(const uint4*)(p + 16);
            o[dt] = __builtin_amdgcn_wmma_f32_16x16x32_bf16(
                        false, ap.v, false, bv.v, (short)0, o[dt], false, false);
        }
    }

    // ---- finalize & store bf16 into attnb[b*T+q][h*128+d] ----
#pragma unroll
    for (int r = 0; r < 8; r++) {
        float inv = 1.0f / l_[r];
        const int qi = qbase + r + hs * 8;
        unsigned short* dst = attnb + (size_t)(b * 2048 + qi) * 2048 + h * 128;
#pragma unroll
        for (int dt = 0; dt < 8; dt++)
            dst[dt * 16 + ml] = f2bf(o[dt][r] * inv);
    }
}

// ---------------------------------------------------------------------------
// Launcher
// ---------------------------------------------------------------------------
extern "C" void kernel_launch(void* const* d_in, const int* in_sizes, int n_in,
                              void* d_out, int out_size, void* d_ws, size_t ws_size,
                              hipStream_t stream) {
    (void)in_sizes; (void)n_in; (void)out_size; (void)ws_size;
    const float* x    = (const float*)d_in[0];
    /* d_in[1] = mask (bool) — causal structure is known, unused */
    const float* cosp = (const float*)d_in[2];
    const float* sinp = (const float*)d_in[3];
    const float* Wq   = (const float*)d_in[4];
    const float* Wk   = (const float*)d_in[5];
    const float* Wv   = (const float*)d_in[6];
    const float* Wo   = (const float*)d_in[7];
    const float* qs   = (const float*)d_in[8];
    const float* ks   = (const float*)d_in[9];

    char* ws = (char*)d_ws;
    unsigned short* xb    = (unsigned short*)(ws);                        // 16 MiB
    unsigned short* Wqkvt = (unsigned short*)(ws + (16ull  << 20));       // 12 MiB
    unsigned short* Wot   = (unsigned short*)(ws + (28ull  << 20));       //  8 MiB
    float*          qkv   = (float*)         (ws + (36ull  << 20));       // 48 MiB
    unsigned short* attnb = (unsigned short*)(ws + (36ull  << 20));       // aliases qkv (dead by then)
    unsigned short* Qr    = (unsigned short*)(ws + (84ull  << 20));       // 16 MiB
    unsigned short* Kr    = (unsigned short*)(ws + (100ull << 20));       //  4 MiB
    unsigned short* Vt    = (unsigned short*)(ws + (104ull << 20));       //  4 MiB
    float* outp = (float*)d_out;

    k_f32_to_bf16 <<<32768, 256, 0, stream>>>(x, xb, 2 * 2048 * 2048);
    k_pack_weights<<<40960, 256, 0, stream>>>(Wq, Wk, Wv, Wo, Wqkvt, Wot);
    // QKV projection: (4096 x 3072) = xb (4096x2048) * Wqkvt^T
    k_gemm_bf16   <<<dim3(3072 / 128, 4096 / 128), 256, 0, stream>>>(
        xb, Wqkvt, qkv, 4096, 3072, 2048);
    k_normrope    <<<10240, 256, 0, stream>>>(qkv, cosp, sinp, qs, ks, Qr, Kr);
    k_vtrans      <<<8192, 256, 0, stream>>>(qkv, Vt);
    k_attention   <<<2 * 16 * 32, 128, 0, stream>>>(Qr, Kr, Vt, attnb);
    // Output projection: (4096 x 2048) = attnb * Wot^T -> f32 d_out
    k_gemm_bf16   <<<dim3(2048 / 128, 4096 / 128), 256, 0, stream>>>(
        attnb, Wot, outp, 4096, 2048, 2048);
}
